// DenseAttention_36206574306106
// MI455X (gfx1250) — matrix operand order repaired and checked by reading, not compiled, not run on hardware
//
#include <hip/hip_runtime.h>

#define B_DIM   16
#define S_LEN   2048
#define D_MODEL 256
#define SCALE   0.0625f   // 1/sqrt(256)

typedef __attribute__((ext_vector_type(16))) _Float16 half16;
typedef __attribute__((ext_vector_type(8)))  _Float16 half8;
typedef __attribute__((ext_vector_type(4)))  _Float16 half4;
typedef __attribute__((ext_vector_type(8)))  float    v8f;
typedef __attribute__((ext_vector_type(4)))  float    v4f;

__device__ __forceinline__ half16 cat16(half8 lo, half8 hi) {
    return __builtin_shufflevector(lo, hi, 0,1,2,3,4,5,6,7,8,9,10,11,12,13,14,15);
}

// 8 consecutive f32 (global or LDS) -> 8 halves (v_cvt_pk_f16_f32 pairs)
__device__ __forceinline__ half8 cvt8(const float* p) {
    const v4f a = *(const v4f*)p;
    const v4f b = *(const v4f*)(p + 4);
    half8 r;
    r[0]=(_Float16)a[0]; r[1]=(_Float16)a[1]; r[2]=(_Float16)a[2]; r[3]=(_Float16)a[3];
    r[4]=(_Float16)b[0]; r[5]=(_Float16)b[1]; r[6]=(_Float16)b[2]; r[7]=(_Float16)b[3];
    return r;
}

// ---------------------------------------------------------------------------
// Pre-pass A: f32 -> f16 copy (for K). 4 elems/thread, fully coalesced.
// ---------------------------------------------------------------------------
__global__ __launch_bounds__(256) void convert_f16_kernel(
        const float* __restrict__ src, _Float16* __restrict__ dst)
{
    const size_t i = ((size_t)blockIdx.x * 256 + threadIdx.x) * 4;
    const v4f x = *(const v4f*)(src + i);
    half4 h;
    h[0]=(_Float16)x[0]; h[1]=(_Float16)x[1]; h[2]=(_Float16)x[2]; h[3]=(_Float16)x[3];
    *(half4*)(dst + i) = h;
}

// ---------------------------------------------------------------------------
// Pre-pass B: V[b][s][d] f32 -> Vt[b][d][s] f16, 32x32 LDS-tiled transpose.
// Grid: B * (S/32) * (D/32) = 8192 blocks of 256 (32x8).
// ---------------------------------------------------------------------------
__global__ __launch_bounds__(256) void transpose_v_kernel(
        const float* __restrict__ V, _Float16* __restrict__ Vt)
{
    const int bid = blockIdx.x;
    const int b   = bid >> 9;          // / 512
    const int r   = bid & 511;
    const int s0  = (r >> 3) * 32;
    const int d0  = (r & 7) * 32;

    __shared__ float tile[32][33];
    const int tx = threadIdx.x & 31;
    const int ty = threadIdx.x >> 5;   // 0..7

    const float* vb = V + (size_t)b * S_LEN * D_MODEL;
#pragma unroll
    for (int i = 0; i < 4; ++i) {
        const int s = ty * 4 + i;
        tile[s][tx] = vb[(size_t)(s0 + s) * D_MODEL + d0 + tx];
    }
    __syncthreads();
    _Float16* vt = Vt + (size_t)b * D_MODEL * S_LEN;
#pragma unroll
    for (int i = 0; i < 4; ++i) {
        const int d = ty * 4 + i;
        vt[(size_t)(d0 + d) * S_LEN + s0 + tx] = (_Float16)tile[tx][d];
    }
}

struct SMem {
    float     w[16 * S_LEN];        // 128 KB: score / weight strip
    _Float16  qh[16 * D_MODEL];     // 8 KB: Q strip in f16
    float     partial[16][17];
    float     rowred[16];
};

// ---------------------------------------------------------------------------
// Fused attention, V_WMMA_F32_16X16X32_F16, f32 accumulate.
// FAST: B operands come pre-converted (Kh) / pre-transposed (Vt) from d_ws.
// One block = one (batch, 16-row q-strip). Grid: 2048 blocks x 256 threads.
// ---------------------------------------------------------------------------
template <bool FAST>
__global__ __launch_bounds__(256) void fused_attn_kernel(
        const float* __restrict__ Q, const float* __restrict__ K,
        const float* __restrict__ V, const int* __restrict__ mask,
        const _Float16* __restrict__ Kh, const _Float16* __restrict__ Vt,
        float* __restrict__ O, float* __restrict__ attnW)
{
    __shared__ SMem sm;

    const int tid  = threadIdx.x;
    const int wave = tid >> 5;
    const int lane = tid & 31;
    const int m = lane & 15;     // A row M / B col N
    const int g = lane >> 4;     // lane group

    const int b  = blockIdx.x >> 7;
    const int q0 = (blockIdx.x & 127) * 16;

    // ---- Stage 0: Q strip (16x256) -> LDS as f16, coalesced ---------------
#pragma unroll
    for (int j = 0; j < 16; ++j) {
        const int i = tid + j * 256;
        sm.qh[i] = (_Float16)Q[(size_t)(b * S_LEN + q0) * D_MODEL + i];
    }
    __syncthreads();

    // ---- Stage 1: 16x2048 score strip into LDS ----------------------------
    // A layout (16-bit A 16x32): K = 16g+e (contiguous)
    half16 afrag[8];
#pragma unroll
    for (int ds = 0; ds < 8; ++ds) {
        const _Float16* qb = sm.qh + m * D_MODEL + ds * 32 + 16 * g;
        afrag[ds] = cat16(*(const half8*)qb, *(const half8*)(qb + 8));
    }
    for (int t = 0; t < 16; ++t) {
        const int k0 = (wave * 16 + t) * 16;
        v8f c = {};
        if (FAST) {
            const _Float16* krow = Kh + (size_t)(b * S_LEN + k0 + m) * D_MODEL + 16 * g;
#pragma unroll
            for (int ds = 0; ds < 8; ++ds) {
                const _Float16* kp = krow + ds * 32;
                half16 bfrag = cat16(*(const half8*)kp, *(const half8*)(kp + 8));
                c = __builtin_amdgcn_wmma_f32_16x16x32_f16(
                        false, afrag[ds], false, bfrag, (short)0, c, false, false);
            }
        } else {
            const float* krow = K + (size_t)(b * S_LEN + k0 + m) * D_MODEL + 16 * g;
#pragma unroll
            for (int ds = 0; ds < 8; ++ds) {
                const float* kp = krow + ds * 32;
                half16 bfrag = cat16(cvt8(kp), cvt8(kp + 8));
                c = __builtin_amdgcn_wmma_f32_16x16x32_f16(
                        false, afrag[ds], false, bfrag, (short)0, c, false, false);
            }
        }
        const int kcol = k0 + m;
#pragma unroll
        for (int i = 0; i < 8; ++i) {
            const int r = i + 8 * g;                       // C layout: M = i + 8g
            const int mv = mask[(size_t)(q0 + r) * S_LEN + kcol];
            sm.w[r * S_LEN + kcol] = (mv == 0) ? -3.0e38f : c[i] * SCALE;
        }
    }
    __syncthreads();

    // ---- Stage 2: softmax over each of the 16 rows ------------------------
    {
        const int r   = tid >> 4;
        const int sub = tid & 15;
        float lmax = -3.4e38f;
        for (int j = 0; j < 128; ++j)
            lmax = fmaxf(lmax, sm.w[r * S_LEN + sub + j * 16]);
        sm.partial[r][sub] = lmax;
        __syncthreads();
        if (tid < 16) {
            float mx = sm.partial[tid][0];
            for (int j2 = 1; j2 < 16; ++j2) mx = fmaxf(mx, sm.partial[tid][j2]);
            sm.rowred[tid] = mx;
        }
        __syncthreads();
        const float mx = sm.rowred[r];
        float lsum = 0.f;
        for (int j = 0; j < 128; ++j) {
            const int idx = r * S_LEN + sub + j * 16;
            const float e = __expf(sm.w[idx] - mx);        // masked -> exactly 0
            sm.w[idx] = e;
            lsum += e;
        }
        sm.partial[r][sub] = lsum;
        __syncthreads();
        if (tid < 16) {
            float s = 0.f;
            for (int j2 = 0; j2 < 16; ++j2) s += sm.partial[tid][j2];
            sm.rowred[tid] = 1.0f / s;
        }
        __syncthreads();

        float* wout = attnW + (size_t)b * S_LEN * S_LEN + (size_t)q0 * S_LEN;
        for (int r2 = 0; r2 < 16; ++r2) {
            const float inv = sm.rowred[r2];
#pragma unroll
            for (int j = 0; j < 8; ++j) {
                const int col = tid + j * 256;
                const float wv = sm.w[r2 * S_LEN + col] * inv;
                sm.w[r2 * S_LEN + col] = wv;
                wout[(size_t)r2 * S_LEN + col] = wv;
            }
        }
    }
    __syncthreads();

    // ---- Stage 3: O strip = W(LDS,f32->f16) @ V ---------------------------
    {
        const int d0a = wave * 32;                 // two 16-wide d-tiles / wave
        const int d0b = wave * 32 + 16;
        v8f c0 = {}, c1 = {};
        if (FAST) {
            // B: Vt[b][d0+n][kk + 16g + e] -> 16 contiguous halves
            const _Float16* vt0 = Vt + (size_t)(b * D_MODEL + d0a + m) * S_LEN + 16 * g;
            const _Float16* vt1 = vt0 + (size_t)16 * S_LEN;
            for (int kk = 0; kk < S_LEN; kk += 32) {
                const float* wp = sm.w + m * S_LEN + kk + 16 * g;
                half16 a  = cat16(cvt8(wp), cvt8(wp + 8));
                half16 b0 = cat16(*(const half8*)(vt0 + kk), *(const half8*)(vt0 + kk + 8));
                half16 b1 = cat16(*(const half8*)(vt1 + kk), *(const half8*)(vt1 + kk + 8));
                c0 = __builtin_amdgcn_wmma_f32_16x16x32_f16(
                        false, a, false, b0, (short)0, c0, false, false);
                c1 = __builtin_amdgcn_wmma_f32_16x16x32_f16(
                        false, a, false, b1, (short)0, c1, false, false);
            }
        } else {
            const float* vb0 = V + (size_t)b * S_LEN * D_MODEL
                                 + (size_t)(16 * g) * D_MODEL + d0a + m;
            const float* vb1 = vb0 + 16;
            for (int kk = 0; kk < S_LEN; kk += 32) {
                const float* wp = sm.w + m * S_LEN + kk + 16 * g;
                half16 a = cat16(cvt8(wp), cvt8(wp + 8));
                half16 b0, b1;
#pragma unroll
                for (int e = 0; e < 16; ++e) {
                    const size_t off = (size_t)(kk + e) * D_MODEL;
                    b0[e] = (_Float16)vb0[off];
                    b1[e] = (_Float16)vb1[off];
                }
                c0 = __builtin_amdgcn_wmma_f32_16x16x32_f16(
                        false, a, false, b0, (short)0, c0, false, false);
                c1 = __builtin_amdgcn_wmma_f32_16x16x32_f16(
                        false, a, false, b1, (short)0, c1, false, false);
            }
        }
        float* orow = O + (size_t)(b * S_LEN + q0) * D_MODEL;
#pragma unroll
        for (int i = 0; i < 8; ++i) {
            const int r = i + 8 * g;
            orow[(size_t)r * D_MODEL + d0a + m] = c0[i];
            orow[(size_t)r * D_MODEL + d0b + m] = c1[i];
        }
    }
}

// ---------------------------------------------------------------------------
extern "C" void kernel_launch(void* const* d_in, const int* in_sizes, int n_in,
                              void* d_out, int out_size, void* d_ws, size_t ws_size,
                              hipStream_t stream)
{
    (void)in_sizes; (void)n_in; (void)out_size;

    const float* q    = (const float*)d_in[0];
    const float* k    = (const float*)d_in[1];
    const float* v    = (const float*)d_in[2];
    const int*   mask = (const int*)d_in[3];

    float* out  = (float*)d_out;                            // [B,S,D]
    float* attn = out + (size_t)B_DIM * S_LEN * D_MODEL;    // [B,S,S]

    const size_t nElem   = (size_t)B_DIM * S_LEN * D_MODEL; // 8.4M
    const size_t khBytes = nElem * sizeof(_Float16);        // 16 MB

    if (ws_size >= 2 * khBytes) {
        _Float16* kh = (_Float16*)d_ws;
        _Float16* vt = (_Float16*)((char*)d_ws + khBytes);
        convert_f16_kernel<<<(int)(nElem / 4 / 256), 256, 0, stream>>>(k, kh);
        transpose_v_kernel<<<B_DIM * (S_LEN / 32) * (D_MODEL / 32), 256, 0, stream>>>(v, vt);
        fused_attn_kernel<true><<<B_DIM * (S_LEN / 16), 256, 0, stream>>>(
            q, k, v, mask, kh, vt, out, attn);
    } else {
        fused_attn_kernel<false><<<B_DIM * (S_LEN / 16), 256, 0, stream>>>(
            q, k, v, mask, nullptr, nullptr, out, attn);
    }
}
